// WTM_15401752723496
// MI455X (gfx1250) — matrix operand order repaired and checked by reading, not compiled, not run on hardware
//
#include <hip/hip_runtime.h>
#include <hip/hip_bf16.h>
#include <hip/hip_fp16.h>

// ---------------------------------------------------------------------------
// CDNA5 (gfx1250) wave32 WMMA types and fragment helpers
// ---------------------------------------------------------------------------
typedef _Float16 v16h __attribute__((ext_vector_type(16)));
typedef _Float16 v8h  __attribute__((ext_vector_type(8)));
typedef float    v8f  __attribute__((ext_vector_type(8)));
typedef unsigned int u32x4 __attribute__((ext_vector_type(4)));
typedef unsigned int u32x8 __attribute__((ext_vector_type(8)));

union U16h { v16h v; v8h h[2]; };

// A-matrix 16x32 f16 fragment (wave32):
//   lane L: m = L&15 ; k(e) = (e<8 ? 0 : 16) + (L>>4)*8 + (e&7)
__device__ __forceinline__ v16h frag_a(const _Float16* base, int lda, int lane) {
    const int m = lane & 15, g = lane >> 4;
    const _Float16* p = base + m * lda + g * 8;
    U16h u;
    u.h[0] = *(const v8h*)(p);        // k = g*8 + 0..7
    u.h[1] = *(const v8h*)(p + 16);   // k = 16 + g*8 + 0..7
    return u.v;
}

// B-matrix 32x16 f16 fragment for X @ W^T with W row-major (out x in):
//   lane L: n = L&15 ; k(e) = (e<8 ? 0 : 16) + (L>>4)*8 + (e&7)
__device__ __forceinline__ v16h frag_b(const _Float16* base, int ldb, int lane) {
    const int n = lane & 15, g = lane >> 4;
    const _Float16* p = base + n * ldb + g * 8;
    U16h u;
    u.h[0] = *(const v8h*)(p);
    u.h[1] = *(const v8h*)(p + 16);
    return u.v;
}

// acc += A(m0:,k) * W^T(k, n0:) over K, 16x16x32 f16 WMMA
__device__ __forceinline__ v8f wmma_tile(v8f acc,
                                         const _Float16* sA, int lda,
                                         const _Float16* W,  int ldw,
                                         int m0, int n0, int K, int lane) {
#pragma unroll
    for (int k0 = 0; k0 < K; k0 += 32) {
        v16h a = frag_a(sA + m0 * lda + k0, lda, lane);
        v16h b = frag_b(W + n0 * ldw + k0, ldw, lane);
        acc = __builtin_amdgcn_wmma_f32_16x16x32_f16(
            false, a, false, b, (short)0, acc, false, false);
    }
    return acc;
}

// ---------------------------------------------------------------------------
// Geometry:  x = (B=4, C=128, H=256, W=256), WS=8, NH=8, hd=16
// window id w in [0,4096): b=w>>10, wy=(w>>5)&31, wx=w&31 ; token n = i*8+j
// ---------------------------------------------------------------------------
#define HW   65536
#define CCH  128
#define LDA  136   // LDS row stride (halves) for 64x128 tiles
#define LDQ  392   // LDS row stride for 64x384 qkv
#define LDC  132   // LDS row stride (floats) for f32 accumulator
#define LDH  520   // LDS row stride for 64x512 LeFF hidden

// -------------------------- weight conversion ------------------------------
__global__ void wtm_cvt(const float* __restrict__ s, _Float16* __restrict__ d, int n) {
    int i = blockIdx.x * blockDim.x + threadIdx.x;
    if (i < n) d[i] = (_Float16)s[i];
}

// ------------------ LayerNorm1 + window partition -> f16 -------------------
__global__ __launch_bounds__(256)
void wtm_ln1(const float* __restrict__ x, const float* __restrict__ gw,
             const float* __restrict__ gb, _Float16* __restrict__ Xn) {
    __shared__ float sT[64][LDC];
    __shared__ float sMu[64], sRs[64];
    const int w = blockIdx.x, t = threadIdx.x;
    const int b = w >> 10, wy = (w >> 5) & 31, wx = w & 31;
    const int j = t & 7, i = (t >> 3) & 7, cg = t >> 6;
    const int p = i * 8 + j;
    const size_t pix = (size_t)(wy * 8 + i) * 256 + (wx * 8 + j);
    const float* base = x + (size_t)b * CCH * HW + pix;
    for (int c = cg * 32; c < cg * 32 + 32; ++c)
        sT[p][c] = base[(size_t)c * HW];
    __syncthreads();
    if (t < 64) {
        float s = 0.f, ss = 0.f;
        for (int c = 0; c < CCH; ++c) { float v = sT[t][c]; s += v; ss += v * v; }
        float mu = s * (1.f / 128.f);
        float var = ss * (1.f / 128.f) - mu * mu;
        sMu[t] = mu; sRs[t] = rsqrtf(var + 1e-5f);
    }
    __syncthreads();
    _Float16* dst = Xn + (size_t)w * 64 * CCH;
    for (int c = cg * 32; c < cg * 32 + 32; ++c)
        dst[p * CCH + c] = (_Float16)((sT[p][c] - sMu[p]) * sRs[p] * gw[c] + gb[c]);
}

// -------- fused reMSA per window: qkv, head-mix attn, proj, conv branch ----
__global__ __launch_bounds__(256)
void wtm_attn(const float* __restrict__ x, const _Float16* __restrict__ Xn,
              const _Float16* __restrict__ Wqkv, const float* __restrict__ qkv_b,
              const float* __restrict__ pos_bias,
              const _Float16* __restrict__ Wout,  const float* __restrict__ out_b,
              const _Float16* __restrict__ Wcp,   const float* __restrict__ cp_b,
              const _Float16* __restrict__ Wpw1,  const float* __restrict__ pw1_b,
              const float* __restrict__ dw1_w,    const float* __restrict__ dw1_b,
              const _Float16* __restrict__ Wtok,  const float* __restrict__ tok_b,
              float* __restrict__ out) {
    __shared__ __attribute__((aligned(16))) unsigned char smem[118784];
    _Float16* sX   = (_Float16*)(smem);                    // 64 x 136 f16
    _Float16* sB   = (_Float16*)(smem + 17408);            // 64 x 392 f16 (qkv)
    _Float16* sT1  = sB;                                   // reuse: 64 x 136
    _Float16* sT2  = (_Float16*)(smem + 17408 + 17408);    // reuse: 64 x 136
    _Float16* sAO  = (_Float16*)(smem + 67584);            // 64 x 136 f16
    float*    sAcc = (float*)   (smem + 84992);            // 64 x 132 f32

    const int w = blockIdx.x, t = threadIdx.x;
    const int wave = t >> 5, lane = t & 31;
    const int ln = lane & 15, lg = lane >> 4;

    // warm L2/WGP$ for the weight B-fragment stream (global_prefetch_b8)
    __builtin_prefetch(Wqkv + (size_t)t * 192, 0, 0);
    __builtin_prefetch(Wout + (size_t)t * 64, 0, 0);
    __builtin_prefetch(Wcp  + (size_t)t * 64, 0, 0);
    __builtin_prefetch(Wpw1 + (size_t)t * 64, 0, 0);
    __builtin_prefetch(Wtok + (size_t)t * 64, 0, 0);

    // stage 1: TDM async-tensor load of the 64x128 f16 window tile into LDS.
    // 2D tile (tile_dim0=128 halves, tile_dim1=64 rows), data_size=2B,
    // LDS padding: pad_interval=5 (every 256B = one row), pad_amount=3 (16B)
    // reproduces the LDA=136 row stride in hardware.
    if (t == 0) {
        unsigned long long ga =
            (unsigned long long)(const void*)(Xn + (size_t)w * 8192);
        u32x4 g0;
        g0.x = 1u;                                   // count=1, user descriptor
        g0.y = (unsigned)(unsigned long long)(size_t)sX;   // lds_addr (offset)
        g0.z = (unsigned)(ga & 0xffffffffu);         // global_addr lo
        g0.w = (unsigned)((ga >> 32) & 0x01ffffffu) | (2u << 30); // hi + type=2
        u32x8 g1;
        g1.s0 = (1u << 16)            // data_size = 2 bytes
              | (1u << 20)            // pad_enable
              | (5u << 22)            // pad_interval: 256 bytes
              | (3u << 25);           // pad_amount: 4 DWORDs = 16 bytes
        g1.s1 = (128u & 0xffffu) << 16;              // tensor_dim0 = 128 (lo16)
        g1.s2 = (64u << 16);                         // tensor_dim1 = 64 (lo16)
        g1.s3 = (128u << 16);                        // tile_dim0 = 128
        g1.s4 = 64u;                                 // tile_dim1 = 64
        g1.s5 = 128u;                                // tensor_dim0_stride lo32
        g1.s6 = 0u;
        g1.s7 = 0u;
        asm volatile("tensor_load_to_lds %0, %1" :: "s"(g0), "s"(g1) : "memory");
        __builtin_amdgcn_s_wait_tensorcnt(0);
    }
    __syncthreads();

    // stage 2: qkv = X @ Wqkv^T + b   (64x384)
    for (int idx = wave; idx < 96; idx += 8) {
        const int m0 = (idx & 3) * 16, n0 = (idx >> 2) * 16;
        v8f acc = {};
        acc = wmma_tile(acc, sX, LDA, Wqkv, CCH, m0, n0, CCH, lane);
        const float bb = qkv_b[n0 + ln];
#pragma unroll
        for (int r = 0; r < 8; ++r)
            sB[(m0 + lg * 8 + r) * LDQ + n0 + ln] = (_Float16)(acc[r] + bb);
    }
    __syncthreads();

    // stage 3: per-token head-mixing attention (8x8 scores, softmax, @v)
    {
        const int n = t >> 2, sub = t & 3;
        const _Float16* row = sB + n * LDQ;   // head h -> [h*48 + {q:0..15, k:16..31, v:32..47}]
#pragma unroll
        for (int hh = sub * 2; hh < sub * 2 + 2; ++hh) {
            float q[16];
#pragma unroll
            for (int d = 0; d < 16; ++d) q[d] = (float)row[hh * 48 + d];
            float s[8], mx = -1e30f;
#pragma unroll
            for (int g = 0; g < 8; ++g) {
                float dot = 0.f;
#pragma unroll
                for (int d = 0; d < 16; ++d) dot += q[d] * (float)row[g * 48 + 16 + d];
                s[g] = dot * 0.25f + pos_bias[(n * 8 + hh) * 8 + g];
                mx = fmaxf(mx, s[g]);
            }
            float sum = 0.f;
#pragma unroll
            for (int g = 0; g < 8; ++g) { s[g] = __expf(s[g] - mx); sum += s[g]; }
            const float inv = 1.f / sum;
            float ao[16];
#pragma unroll
            for (int d = 0; d < 16; ++d) ao[d] = 0.f;
#pragma unroll
            for (int g = 0; g < 8; ++g) {
                const float pg = s[g] * inv;
#pragma unroll
                for (int d = 0; d < 16; ++d) ao[d] += pg * (float)row[g * 48 + 32 + d];
            }
#pragma unroll
            for (int d = 0; d < 16; ++d) sAO[n * LDA + hh * 16 + d] = (_Float16)ao[d];
        }
    }
    __syncthreads();

    // stage 4: O = AO @ Wout^T + b  -> f32 accumulator
    for (int idx = wave; idx < 32; idx += 8) {
        const int m0 = (idx & 3) * 16, n0 = (idx >> 2) * 16;
        v8f acc = {};
        acc = wmma_tile(acc, sAO, LDA, Wout, CCH, m0, n0, CCH, lane);
        const float bb = out_b[n0 + ln];
#pragma unroll
        for (int r = 0; r < 8; ++r)
            sAcc[(m0 + lg * 8 + r) * LDC + n0 + ln] = acc[r] + bb;
    }
    // stage 5: conv branch xc1 = X @ Wconvp^T + b  (overwrites qkv region)
    for (int idx = wave; idx < 32; idx += 8) {
        const int m0 = (idx & 3) * 16, n0 = (idx >> 2) * 16;
        v8f acc = {};
        acc = wmma_tile(acc, sX, LDA, Wcp, CCH, m0, n0, CCH, lane);
        const float bb = cp_b[n0 + ln];
#pragma unroll
        for (int r = 0; r < 8; ++r)
            sT1[(m0 + lg * 8 + r) * LDA + n0 + ln] = (_Float16)(acc[r] + bb);
    }
    __syncthreads();

    // stage 6: xc2 = xc1 @ Wpw1^T + b
    for (int idx = wave; idx < 32; idx += 8) {
        const int m0 = (idx & 3) * 16, n0 = (idx >> 2) * 16;
        v8f acc = {};
        acc = wmma_tile(acc, sT1, LDA, Wpw1, CCH, m0, n0, CCH, lane);
        const float bb = pw1_b[n0 + ln];
#pragma unroll
        for (int r = 0; r < 8; ++r)
            sT2[(m0 + lg * 8 + r) * LDA + n0 + ln] = (_Float16)(acc[r] + bb);
    }
    __syncthreads();

    // stage 7: depthwise conv1d (k=3, pad=1) along token axis
    {
        const int p = t >> 2, cb = (t & 3) * 32;
        for (int c = cb; c < cb + 32; ++c) {
            float v  = dw1_b[c];
            float xm = (p > 0)  ? (float)sT2[(p - 1) * LDA + c] : 0.f;
            float x0 = (float)sT2[p * LDA + c];
            float xp = (p < 63) ? (float)sT2[(p + 1) * LDA + c] : 0.f;
            v += dw1_w[c * 3 + 0] * xm + dw1_w[c * 3 + 1] * x0 + dw1_w[c * 3 + 2] * xp;
            sAO[p * LDA + c] = (_Float16)v;   // sAO reused as z
        }
    }
    __syncthreads();

    // stage 8: acc += z @ Wtok^T + b  (same tile ownership as stage 4)
    for (int idx = wave; idx < 32; idx += 8) {
        const int m0 = (idx & 3) * 16, n0 = (idx >> 2) * 16;
        v8f acc;
#pragma unroll
        for (int r = 0; r < 8; ++r)
            acc[r] = sAcc[(m0 + lg * 8 + r) * LDC + n0 + ln];
        acc = wmma_tile(acc, sAO, LDA, Wtok, CCH, m0, n0, CCH, lane);
        const float bb = tok_b[n0 + ln];
#pragma unroll
        for (int r = 0; r < 8; ++r)
            sAcc[(m0 + lg * 8 + r) * LDC + n0 + ln] = acc[r] + bb;
    }
    __syncthreads();

    // stage 9: window merge + residual -> out (mid)
    {
        const int b = w >> 10, wy = (w >> 5) & 31, wx = w & 31;
        const int j = t & 7, i = (t >> 3) & 7, cg = t >> 6;
        const int p = i * 8 + j;
        const size_t pix = (size_t)(wy * 8 + i) * 256 + (wx * 8 + j);
        const float* xin = x + (size_t)b * CCH * HW + pix;
        float* po = out + (size_t)b * CCH * HW + pix;
        for (int c = cg * 32; c < cg * 32 + 32; ++c)
            po[(size_t)c * HW] = xin[(size_t)c * HW] + sAcc[p * LDC + c];
    }
}

// ------------- LayerNorm2 + 1x1 conv (lpw) -> pixel-major f16 --------------
__global__ __launch_bounds__(256)
void wtm_ln2_lpw(const float* __restrict__ mid, const float* __restrict__ gw,
                 const float* __restrict__ gb, const _Float16* __restrict__ Wlpw,
                 _Float16* __restrict__ Y2) {
    __shared__ float sT[64][LDC];
    __shared__ float sMu[64], sRs[64];
    __shared__ __attribute__((aligned(16))) _Float16 sXn[64][LDA];
    const int w = blockIdx.x, t = threadIdx.x;
    const int b = w >> 10, wy = (w >> 5) & 31, wx = w & 31;
    const int j = t & 7, i = (t >> 3) & 7, cg = t >> 6;
    const int p = i * 8 + j;
    const size_t pix = (size_t)(wy * 8 + i) * 256 + (wx * 8 + j);
    const float* base = mid + (size_t)b * CCH * HW + pix;
    __builtin_prefetch(Wlpw + (size_t)t * 64, 0, 0);
    for (int c = cg * 32; c < cg * 32 + 32; ++c)
        sT[p][c] = base[(size_t)c * HW];
    __syncthreads();
    if (t < 64) {
        float s = 0.f, ss = 0.f;
        for (int c = 0; c < CCH; ++c) { float v = sT[t][c]; s += v; ss += v * v; }
        float mu = s * (1.f / 128.f);
        float var = ss * (1.f / 128.f) - mu * mu;
        sMu[t] = mu; sRs[t] = rsqrtf(var + 1e-5f);
    }
    __syncthreads();
    for (int c = cg * 32; c < cg * 32 + 32; ++c)
        sXn[p][c] = (_Float16)((sT[p][c] - sMu[p]) * sRs[p] * gw[c] + gb[c]);
    __syncthreads();

    const int wave = t >> 5, lane = t & 31, ln = lane & 15, lg = lane >> 4;
    for (int idx = wave; idx < 32; idx += 8) {
        const int m0 = (idx & 3) * 16, n0 = (idx >> 2) * 16;
        v8f acc = {};
        acc = wmma_tile(acc, &sXn[0][0], LDA, Wlpw, CCH, m0, n0, CCH, lane);
#pragma unroll
        for (int r = 0; r < 8; ++r) {
            const int pp = m0 + lg * 8 + r, ii = pp >> 3, jj = pp & 7;
            const size_t gp = (size_t)b * HW + (size_t)(wy * 8 + ii) * 256 + (wx * 8 + jj);
            Y2[gp * CCH + n0 + ln] = (_Float16)acc[r];   // 1x1 conv has no bias
        }
    }
}

// ----- LeFF: 3x3 depthwise (with halo) + fc1/gelu + fc2 + residual --------
__global__ __launch_bounds__(256)
void wtm_leff(const _Float16* __restrict__ Y2,
              const float* __restrict__ ldw_w, const float* __restrict__ ldw_b,
              const _Float16* __restrict__ Wfc1, const float* __restrict__ fc1_b,
              const _Float16* __restrict__ Wfc2, const float* __restrict__ fc2_b,
              float* __restrict__ out) {
    __shared__ __attribute__((aligned(16))) unsigned char smem[83968];
    _Float16* sZ = (_Float16*)(smem);           // 64 x 136
    _Float16* sH = (_Float16*)(smem + 17408);   // 64 x 520
    const int w = blockIdx.x, t = threadIdx.x;
    const int b = w >> 10, wy = (w >> 5) & 31, wx = w & 31;

    __builtin_prefetch(Wfc1 + (size_t)t * 256, 0, 0);
    __builtin_prefetch(Wfc2 + (size_t)t * 256, 0, 0);

    // depthwise 3x3 with zero padding (halo from neighboring tiles via global)
    {
        const int p = t >> 2, cb = (t & 3) * 32;
        const int ii = p >> 3, jj = p & 7;
        const int hh = wy * 8 + ii, ww = wx * 8 + jj;
        float acc[32];
#pragma unroll
        for (int c = 0; c < 32; ++c) acc[c] = ldw_b[cb + c];
        for (int dy = -1; dy <= 1; ++dy)
            for (int dx = -1; dx <= 1; ++dx) {
                const int yy = hh + dy, xx = ww + dx;
                if (yy < 0 || yy >= 256 || xx < 0 || xx >= 256) continue;
                const _Float16* src =
                    Y2 + ((size_t)b * HW + (size_t)yy * 256 + xx) * CCH + cb;
                const int kk = (dy + 1) * 3 + (dx + 1);
#pragma unroll
                for (int c = 0; c < 32; ++c)
                    acc[c] += ldw_w[(cb + c) * 9 + kk] * (float)src[c];
            }
#pragma unroll
        for (int c = 0; c < 32; ++c) sZ[p * LDA + cb + c] = (_Float16)acc[c];
    }
    __syncthreads();

    const int wave = t >> 5, lane = t & 31, ln = lane & 15, lg = lane >> 4;

    // h = gelu(z @ fc1^T + b)   (64 x 512)
    for (int idx = wave; idx < 128; idx += 8) {
        const int m0 = (idx & 3) * 16, n0 = (idx >> 2) * 16;
        v8f acc = {};
        acc = wmma_tile(acc, sZ, LDA, Wfc1, CCH, m0, n0, CCH, lane);
        const float bb = fc1_b[n0 + ln];
#pragma unroll
        for (int r = 0; r < 8; ++r) {
            const float vv = acc[r] + bb;
            const float ge = 0.5f * vv * (1.f + erff(vv * 0.70710678f));
            sH[(m0 + lg * 8 + r) * LDH + n0 + ln] = (_Float16)ge;
        }
    }
    __syncthreads();

    // out += h @ fc2^T + b   (64 x 128, K = 512)
    for (int idx = wave; idx < 32; idx += 8) {
        const int m0 = (idx & 3) * 16, n0 = (idx >> 2) * 16;
        v8f acc = {};
        acc = wmma_tile(acc, sH, LDH, Wfc2, 512, m0, n0, 512, lane);
        const float bb = fc2_b[n0 + ln];
#pragma unroll
        for (int r = 0; r < 8; ++r) {
            const int pp = m0 + lg * 8 + r, ii = pp >> 3, jj = pp & 7;
            const size_t oi = ((size_t)b * CCH + n0 + ln) * HW +
                              (size_t)(wy * 8 + ii) * 256 + (wx * 8 + jj);
            out[oi] = out[oi] + (acc[r] + bb);
        }
    }
}

// ---------------------------------------------------------------------------
extern "C" void kernel_launch(void* const* d_in, const int* in_sizes, int n_in,
                              void* d_out, int out_size, void* d_ws, size_t ws_size,
                              hipStream_t stream) {
    (void)in_sizes; (void)n_in; (void)out_size; (void)ws_size;
    const float* x       = (const float*)d_in[0];
    const float* n1w     = (const float*)d_in[1];
    const float* n1b     = (const float*)d_in[2];
    const float* qkv_w   = (const float*)d_in[3];
    const float* qkv_b   = (const float*)d_in[4];
    const float* pos_b   = (const float*)d_in[5];
    const float* out_w   = (const float*)d_in[6];
    const float* out_b   = (const float*)d_in[7];
    const float* convp_w = (const float*)d_in[8];
    const float* convp_b = (const float*)d_in[9];
    const float* pw1_w   = (const float*)d_in[10];
    const float* pw1_b   = (const float*)d_in[11];
    const float* dw1_w   = (const float*)d_in[12];
    const float* dw1_b   = (const float*)d_in[13];
    const float* tok_w   = (const float*)d_in[14];
    const float* tok_b   = (const float*)d_in[15];
    const float* n2w     = (const float*)d_in[16];
    const float* n2b     = (const float*)d_in[17];
    const float* lpw_w   = (const float*)d_in[18];
    const float* ldw_w   = (const float*)d_in[19];
    const float* ldw_b   = (const float*)d_in[20];
    const float* fc1_w   = (const float*)d_in[21];
    const float* fc1_b   = (const float*)d_in[22];
    const float* fc2_w   = (const float*)d_in[23];
    const float* fc2_b   = (const float*)d_in[24];
    float* outp = (float*)d_out;

    // workspace layout (halves)
    _Float16* wb     = (_Float16*)d_ws;
    _Float16* Wqkv   = wb;            // 49152
    _Float16* Wout   = wb + 49152;    // 16384
    _Float16* Wconvp = wb + 65536;    // 16384
    _Float16* Wpw1   = wb + 81920;    // 16384
    _Float16* Wtok   = wb + 98304;    // 16384
    _Float16* Wlpw   = wb + 114688;   // 16384
    _Float16* Wfc1   = wb + 131072;   // 65536
    _Float16* Wfc2   = wb + 196608;   // 65536
    _Float16* Xn     = wb + 262144;   // 4096*64*128 halves; reused later as Y2

    auto cvt = [&](const float* s, _Float16* d, int n) {
        wtm_cvt<<<(n + 255) / 256, 256, 0, stream>>>(s, d, n);
    };
    cvt(qkv_w,   Wqkv,   49152);
    cvt(out_w,   Wout,   16384);
    cvt(convp_w, Wconvp, 16384);
    cvt(pw1_w,   Wpw1,   16384);
    cvt(tok_w,   Wtok,   16384);
    cvt(lpw_w,   Wlpw,   16384);
    cvt(fc1_w,   Wfc1,   65536);
    cvt(fc2_w,   Wfc2,   65536);

    wtm_ln1<<<4096, 256, 0, stream>>>(x, n1w, n1b, Xn);
    wtm_attn<<<4096, 256, 0, stream>>>(x, Xn, Wqkv, qkv_b, pos_b, Wout, out_b,
                                       Wconvp, convp_b, Wpw1, pw1_b,
                                       dw1_w, dw1_b, Wtok, tok_b, outp);
    wtm_ln2_lpw<<<4096, 256, 0, stream>>>(outp, n2w, n2b, Wlpw, Xn /* -> Y2 */);
    wtm_leff<<<4096, 256, 0, stream>>>(Xn /* Y2 */, ldw_w, ldw_b,
                                       Wfc1, fc1_b, Wfc2, fc2_b, outp);
}
